// SlidingWindowAttention_67259187855411
// MI455X (gfx1250) — matrix operand order repaired and checked by reading, not compiled, not run on hardware
//
#include <hip/hip_runtime.h>
#include <cstdint>
#include <cstddef>

// ---------------------------------------------------------------------------
// Types for CDNA5 WMMA (wave32): v_wmma_f32_16x16x32_bf16
// ---------------------------------------------------------------------------
typedef __attribute__((ext_vector_type(16))) __bf16       v16bf;
typedef __attribute__((ext_vector_type(8)))  float        v8f;
typedef __attribute__((ext_vector_type(4)))  unsigned int v4u;

union FragBF {
    v16bf v;
    v4u   u[2];
};

__device__ __forceinline__ unsigned short f2bf(float f) {
    unsigned int u = __float_as_uint(f);
    u += 0x7FFFu + ((u >> 16) & 1u);      // round-to-nearest-even
    return (unsigned short)(u >> 16);
}
__device__ __forceinline__ float bf2f(unsigned short h) {
    return __uint_as_float(((unsigned int)h) << 16);
}

__device__ __forceinline__ v8f wmma_bf16(const FragBF& a, const FragBF& b, v8f c) {
    return __builtin_amdgcn_wmma_f32_16x16x32_bf16(
        /*neg_a=*/false, a.v, /*neg_b=*/false, b.v,
        /*c_mod=*/(short)0, c, /*reuse_a=*/false, /*reuse_b=*/false);
}

// CDNA5 async global->LDS copy (ASYNCcnt-tracked, see CDNA5 ISA)
__device__ __forceinline__ void async_copy_b128(unsigned lds_byte_addr,
                                                const void* gaddr) {
    asm volatile("global_load_async_to_lds_b128 %0, %1, off"
                 :: "v"(lds_byte_addr),
                    "v"((unsigned long long)(size_t)gaddr)
                 : "memory");
}
__device__ __forceinline__ void wait_async() {
    asm volatile("s_wait_asynccnt 0x0" ::: "memory");
}

// Problem constants
static constexpr int Bc  = 2;
static constexpr int Sc  = 4096;
static constexpr int Hc  = 16;
static constexpr int KHc = 4;
static constexpr int HDc = 128;
static constexpr int Wc  = 512;
static constexpr int Dc  = 2048;   // H*HD
static constexpr int KVD = 512;    // KH*HD

// ---------------------------------------------------------------------------
// Kernel 1: RMSNorm (fp32 in) -> bf16 out.  One block per token row.
// ---------------------------------------------------------------------------
__global__ __launch_bounds__(256) void rmsnorm_bf16_kernel(
    const float* __restrict__ x, const float* __restrict__ w,
    unsigned short* __restrict__ out)
{
    __shared__ float red[8];
    const int row = blockIdx.x;
    const int tid = threadIdx.x;

    const float* xr = x + (size_t)row * Dc + tid * 8;
    float4 v0 = ((const float4*)xr)[0];
    float4 v1 = ((const float4*)xr)[1];
    float s = v0.x*v0.x + v0.y*v0.y + v0.z*v0.z + v0.w*v0.w
            + v1.x*v1.x + v1.y*v1.y + v1.z*v1.z + v1.w*v1.w;
    #pragma unroll
    for (int m = 1; m < 32; m <<= 1) s += __shfl_xor(s, m, 32);
    if ((tid & 31) == 0) red[tid >> 5] = s;
    __syncthreads();
    float tot = 0.f;
    #pragma unroll
    for (int i = 0; i < 8; ++i) tot += red[i];
    const float rs = rsqrtf(tot / (float)Dc + 1e-6f);

    const float* wr = w + tid * 8;
    unsigned short* o = out + (size_t)row * Dc + tid * 8;
    const float vv[8] = {v0.x, v0.y, v0.z, v0.w, v1.x, v1.y, v1.z, v1.w};
    #pragma unroll
    for (int j = 0; j < 8; ++j) o[j] = f2bf(vv[j] * rs * wr[j]);
}

// ---------------------------------------------------------------------------
// Kernel 2: fused transpose + fp32->bf16 for weights: src[K][N] -> dst[N][K].
// LDS-tiled 32x32 so both global streams stay coalesced.
// ---------------------------------------------------------------------------
__global__ __launch_bounds__(256) void transpose_f32_to_bf16_kernel(
    const float* __restrict__ src, unsigned short* __restrict__ dst, int K, int N)
{
    __shared__ float tile[32][33];
    const int k0 = blockIdx.x * 32;
    const int n0 = blockIdx.y * 32;
    const int tx = threadIdx.x & 31;
    const int ty = threadIdx.x >> 5;       // 0..7
    #pragma unroll
    for (int i = 0; i < 4; ++i)
        tile[ty + i * 8][tx] = src[(size_t)(k0 + ty + i * 8) * N + n0 + tx];
    __syncthreads();
    #pragma unroll
    for (int i = 0; i < 4; ++i)
        dst[(size_t)(n0 + ty + i * 8) * K + k0 + tx] = f2bf(tile[tx][ty + i * 8]);
}

// ---------------------------------------------------------------------------
// Kernel 3: bf16 GEMM  C[MxN] = A[MxK] @ Bt[NxK]^T  (fp32 accum)
// Block tile 128x128, 8 waves; wave (wm,wn) owns 32x64 = 2x4 WMMA tiles.
// Both A and Bt tiles are straight row-major [row][32] copies staged with
// global_load_async_to_lds_b128, double-buffered: tile i+1 streams in while
// WMMAs consume tile i.  LDS stride 40 keeps every frag read a b128.
// ---------------------------------------------------------------------------
__global__ __launch_bounds__(256) void gemm_bf16_kernel(
    const unsigned short* __restrict__ A,   // [M][K]
    const unsigned short* __restrict__ Bt,  // [N][K]  (pre-transposed weights)
    float* __restrict__ Cf, unsigned short* __restrict__ Cb,
    int M, int N, int K)
{
    constexpr int LD = 40;
    __shared__ unsigned short As[2][128 * LD];
    __shared__ unsigned short Bs[2][128 * LD];

    const int tid  = threadIdx.x;
    const int lane = tid & 31;
    const int wave = tid >> 5;
    const int wm   = wave >> 1;     // 0..3 -> 32-row slice
    const int wn   = wave & 1;      // 0..1 -> 64-col slice
    const int hl   = lane >> 4;
    const int lan  = lane & 15;
    const int bm   = blockIdx.x * 128;
    const int bn   = blockIdx.y * 128;

    // staging geometry: chunk c = (row = c>>2, col = (c&3)*8); thread owns
    // chunks {tid, tid+256} of each tile (rows row0 and row0+64).
    const int row0 = tid >> 2;
    const int col0 = (tid & 3) << 3;
    const unsigned short* gA = A  + (size_t)(bm + row0) * K + col0;
    const unsigned short* gB = Bt + (size_t)(bn + row0) * K + col0;
    const size_t   gRowStep  = (size_t)64 * K;       // +64 rows (elements)
    const unsigned ldsRowStep = 64 * LD * 2;         // +64 rows (bytes)

    unsigned ldsA[2], ldsB[2];
    #pragma unroll
    for (int b2 = 0; b2 < 2; ++b2) {
        ldsA[b2] = (unsigned)(size_t)&As[b2][row0 * LD + col0];
        ldsB[b2] = (unsigned)(size_t)&Bs[b2][row0 * LD + col0];
    }

    v8f acc[2][4];
    #pragma unroll
    for (int i = 0; i < 2; ++i)
        #pragma unroll
        for (int j = 0; j < 4; ++j)
            #pragma unroll
            for (int r = 0; r < 8; ++r) acc[i][j][r] = 0.f;

    auto issue_tile = [&](int buf, int k0) {
        const unsigned short* a0 = gA + k0;
        const unsigned short* b0 = gB + k0;
        async_copy_b128(ldsA[buf],              a0);
        async_copy_b128(ldsA[buf] + ldsRowStep, a0 + gRowStep);
        async_copy_b128(ldsB[buf],              b0);
        async_copy_b128(ldsB[buf] + ldsRowStep, b0 + gRowStep);
    };

    const int nTiles = K >> 5;
    issue_tile(0, 0);

    for (int it = 0; it < nTiles; ++it) {
        const int buf = it & 1;
        wait_async();                    // own wave's tile `it` landed in LDS
        __syncthreads();                 // everyone's tile landed; prev compute done
        if (it + 1 < nTiles) issue_tile(buf ^ 1, (it + 1) << 5);

        const unsigned short* pa = As[buf];
        const unsigned short* pb = Bs[buf];
        FragBF a[2];
        #pragma unroll
        for (int mt = 0; mt < 2; ++mt) {
            int m = wm * 32 + mt * 16 + lan;
            a[mt].u[0] = *(const v4u*)&pa[m * LD + hl * 8];
            a[mt].u[1] = *(const v4u*)&pa[m * LD + hl * 8 + 16];
        }
        #pragma unroll
        for (int nt = 0; nt < 4; ++nt) {
            int n = wn * 64 + nt * 16 + lan;
            FragBF bfr;
            bfr.u[0] = *(const v4u*)&pb[n * LD + hl * 8];
            bfr.u[1] = *(const v4u*)&pb[n * LD + hl * 8 + 16];
            #pragma unroll
            for (int mt = 0; mt < 2; ++mt)
                acc[mt][nt] = wmma_bf16(a[mt], bfr, acc[mt][nt]);
        }
    }

    // ---- store C (16x16 f32 C-layout: row = r + hl*8, col = lan)
    if (Cf) {
        #pragma unroll
        for (int mt = 0; mt < 2; ++mt)
            #pragma unroll
            for (int nt = 0; nt < 4; ++nt)
                #pragma unroll
                for (int r = 0; r < 8; ++r) {
                    int row = bm + wm * 32 + mt * 16 + hl * 8 + r;
                    int col = bn + wn * 64 + nt * 16 + lan;
                    Cf[(size_t)row * N + col] = acc[mt][nt][r];
                }
    } else {
        #pragma unroll
        for (int mt = 0; mt < 2; ++mt)
            #pragma unroll
            for (int nt = 0; nt < 4; ++nt)
                #pragma unroll
                for (int r = 0; r < 8; ++r) {
                    int row = bm + wm * 32 + mt * 16 + hl * 8 + r;
                    int col = bn + wn * 64 + nt * 16 + lan;
                    Cb[(size_t)row * N + col] = f2bf(acc[mt][nt][r]);
                }
    }
}

// ---------------------------------------------------------------------------
// Kernel 4: RoPE, in-place on bf16 [B,S,Hn,128]; one thread per (even,odd) pair
// ---------------------------------------------------------------------------
__global__ __launch_bounds__(256) void rope_bf16_kernel(
    unsigned short* __restrict__ t, int Hn, int total_pairs)
{
    int p = blockIdx.x * 256 + threadIdx.x;
    if (p >= total_pairs) return;
    int i   = p & 63;        // frequency index
    int rem = p >> 6;        // (b*S + s)*Hn + h
    int s   = (rem / Hn) % Sc;
    float inv = __powf(10000.f, -(float)(2 * i) / (float)HDc);
    float ang = (float)s * inv;
    float sn, cs;
    __sincosf(ang, &sn, &cs);
    size_t base = (size_t)rem * HDc + 2 * i;
    float te = bf2f(t[base]);
    float to = bf2f(t[base + 1]);
    t[base]     = f2bf(te * cs - to * sn);
    t[base + 1] = f2bf(te * sn + to * cs);
}

// ---------------------------------------------------------------------------
// Kernel 5: sliding-window flash attention.
// Block = (b, h, block n, 128-query tile). 8 waves x 16 q-rows.
// Key chunks of 32 over kj in [q0, q0+639] (exact band coverage).
// Q tile and K chunks staged via async-to-LDS; V staged through VGPRs
// (needs the [hd][key] transpose for the P@V B-fragments).
// ---------------------------------------------------------------------------
__global__ __launch_bounds__(256) void swa_attn_kernel(
    const unsigned short* __restrict__ Q,   // [B,S,16,128] bf16 (rope'd)
    const unsigned short* __restrict__ Kb,  // [B,S,4,128]
    const unsigned short* __restrict__ Vb,  // [B,S,4,128]
    unsigned short* __restrict__ Out)       // [B,S,2048] bf16
{
    constexpr int LQ = 136, LK = 136, LV = 40, LP = 40;
    __shared__ unsigned short Qs[128 * LQ];          // [qrow][hd]
    __shared__ unsigned short Ks[32 * LK];           // [key][hd]  (== B^T layout)
    __shared__ unsigned short Vs[128 * LV];          // [hd][key]  (transposed)
    __shared__ unsigned short Ps[8 * 16 * LP];       // per-wave P tile [16][keys]

    const int tid  = threadIdx.x;
    const int lane = tid & 31;
    const int wave = tid >> 5;
    const int hl   = lane >> 4;
    const int lan  = lane & 15;

    const int bx  = blockIdx.x;
    const int qt  = bx & 3;
    const int nb  = (bx >> 2) & 7;
    const int h   = (bx >> 5) & 15;
    const int b   = bx >> 9;
    const int kvh = h >> 2;

    const int q0  = qt * 128;               // query offset inside the 512-block
    const int sq0 = nb * Wc + q0;           // absolute sequence position

    // ---- Q tile (128x128 bf16) via async-to-LDS: 8 b128 chunks per thread
    {
        const int row0 = tid >> 4;                // 0..15, step 16
        const int col  = (tid & 15) << 3;
        const unsigned short* gq =
            Q + (((size_t)b * Sc + sq0 + row0) * Hc + h) * HDc + col;
        unsigned lq = (unsigned)(size_t)&Qs[row0 * LQ + col];
        #pragma unroll
        for (int i = 0; i < 8; ++i) {
            async_copy_b128(lq, gq);
            gq += (size_t)16 * Hc * HDc;
            lq += 16 * LQ * 2;
        }
    }

    // ---- per-thread K/V staging geometry (2 chunks: keys keyt, keyt+16)
    const int keyt = tid >> 4;                    // 0..15
    const int colt = (tid & 15) << 3;             // 0..120
    const unsigned short* kBase =
        Kb + ((size_t)b * Sc * KHc + kvh) * HDc + colt;
    const unsigned short* vBase =
        Vb + ((size_t)b * Sc * KHc + kvh) * HDc + colt;
    const unsigned ldsK0 = (unsigned)(size_t)&Ks[keyt * LK + colt];
    const unsigned ldsK1 = (unsigned)(size_t)&Ks[(keyt + 16) * LK + colt];
    const int skb = nb * Wc - Wc;                 // window start (may be <0)

    v8f o[8];
    #pragma unroll
    for (int t = 0; t < 8; ++t)
        #pragma unroll
        for (int r = 0; r < 8; ++r) o[t][r] = 0.f;
    float mrow[8], lrow[8];
    #pragma unroll
    for (int r = 0; r < 8; ++r) { mrow[r] = -1e30f; lrow[r] = 0.f; }

    const float scale = 0.088388347648318447f;   // 1/sqrt(128)

    for (int c0 = q0; c0 < q0 + 640; c0 += 32) {
        __syncthreads();          // previous chunk's compute done
        // clamp padding keys (sk<0) to a valid row; the band mask kills them
        int sk0 = skb + c0 + keyt;      sk0 = sk0 < 0 ? 0 : sk0;
        int sk1 = skb + c0 + keyt + 16; sk1 = sk1 < 0 ? 0 : sk1;
        async_copy_b128(ldsK0, kBase + (size_t)sk0 * KVD);
        async_copy_b128(ldsK1, kBase + (size_t)sk1 * KVD);
        v4u zv0 = *(const v4u*)(vBase + (size_t)sk0 * KVD);
        v4u zv1 = *(const v4u*)(vBase + (size_t)sk1 * KVD);
        const unsigned short* v0p = (const unsigned short*)&zv0;
        const unsigned short* v1p = (const unsigned short*)&zv1;
        #pragma unroll
        for (int j = 0; j < 8; ++j) {
            Vs[(colt + j) * LV + keyt]      = v0p[j];
            Vs[(colt + j) * LV + keyt + 16] = v1p[j];
        }
        wait_async();
        __syncthreads();

        // ---- S = Q @ K^T : 16 rows x 32 keys (2 tiles), k-dim 128 in 4 steps
        v8f sacc[2];
        #pragma unroll
        for (int t = 0; t < 2; ++t)
            #pragma unroll
            for (int r = 0; r < 8; ++r) sacc[t][r] = 0.f;

        #pragma unroll
        for (int ks = 0; ks < 128; ks += 32) {
            FragBF a;
            int m = wave * 16 + lan;
            a.u[0] = *(const v4u*)&Qs[m * LQ + ks + hl * 8];
            a.u[1] = *(const v4u*)&Qs[m * LQ + ks + hl * 8 + 16];
            #pragma unroll
            for (int t = 0; t < 2; ++t) {
                FragBF bfr;
                int n = t * 16 + lan;
                bfr.u[0] = *(const v4u*)&Ks[n * LK + ks + hl * 8];
                bfr.u[1] = *(const v4u*)&Ks[n * LK + ks + hl * 8 + 16];
                sacc[t] = wmma_bf16(a, bfr, sacc[t]);
            }
        }

        // ---- mask + scale + chunk row-max (16-lane butterfly)
        float cmax[8];
        #pragma unroll
        for (int r = 0; r < 8; ++r) {
            int qi = q0 + wave * 16 + hl * 8 + r;      // in-block query index
            float best = -1e30f;
            #pragma unroll
            for (int t = 0; t < 2; ++t) {
                int kj = c0 + t * 16 + lan;            // in-window key index
                float val = sacc[t][r] * scale;
                bool valid = (kj >= qi) && (kj <= qi + Wc) && (nb >= 1 || kj >= Wc);
                val = valid ? val : -1e30f;
                sacc[t][r] = val;
                best = fmaxf(best, val);
            }
            #pragma unroll
            for (int msk = 1; msk < 16; msk <<= 1)
                best = fmaxf(best, __shfl_xor(best, msk, 32));
            cmax[r] = best;
        }

        // ---- online softmax update + rescale O
        #pragma unroll
        for (int r = 0; r < 8; ++r) {
            float mnew  = fmaxf(mrow[r], cmax[r]);
            float alpha = __expf(mrow[r] - mnew);
            mrow[r] = mnew;
            float psum = 0.f;
            #pragma unroll
            for (int t = 0; t < 2; ++t) {
                float p = __expf(sacc[t][r] - mnew);
                sacc[t][r] = p;
                psum += p;
            }
            #pragma unroll
            for (int msk = 1; msk < 16; msk <<= 1)
                psum += __shfl_xor(psum, msk, 32);
            lrow[r] = lrow[r] * alpha + psum;
            #pragma unroll
            for (int t = 0; t < 8; ++t) o[t][r] *= alpha;
        }

        // ---- P -> wave-private LDS (C-layout -> A-layout round trip)
        unsigned short* pw = &Ps[wave * 16 * LP];
        #pragma unroll
        for (int r = 0; r < 8; ++r)
            #pragma unroll
            for (int t = 0; t < 2; ++t)
                pw[(hl * 8 + r) * LP + t * 16 + lan] = f2bf(sacc[t][r]);

        // ---- O += P @ V  (k = 32 keys, 8 hd tiles)
        {
            FragBF a;
            a.u[0] = *(const v4u*)&pw[lan * LP + hl * 8];
            a.u[1] = *(const v4u*)&pw[lan * LP + hl * 8 + 16];
            #pragma unroll
            for (int t = 0; t < 8; ++t) {
                FragBF bfr;
                int n = t * 16 + lan;
                bfr.u[0] = *(const v4u*)&Vs[n * LV + hl * 8];
                bfr.u[1] = *(const v4u*)&Vs[n * LV + hl * 8 + 16];
                o[t] = wmma_bf16(a, bfr, o[t]);
            }
        }
    }

    // ---- epilogue: O /= l, store bf16 to [B,S, h*128 + hd]
    #pragma unroll
    for (int r = 0; r < 8; ++r) {
        float inv = 1.0f / lrow[r];
        size_t srow = (size_t)b * Sc + sq0 + wave * 16 + hl * 8 + r;
        #pragma unroll
        for (int t = 0; t < 8; ++t)
            Out[srow * Dc + h * HDc + t * 16 + lan] = f2bf(o[t][r] * inv);
    }
}

// ---------------------------------------------------------------------------
// Host orchestration
// ---------------------------------------------------------------------------
extern "C" void kernel_launch(void* const* d_in, const int* in_sizes, int n_in,
                              void* d_out, int out_size, void* d_ws, size_t ws_size,
                              hipStream_t stream)
{
    const float* x    = (const float*)d_in[0];
    const float* ln_w = (const float*)d_in[1];
    const float* wq   = (const float*)d_in[2];
    const float* wk   = (const float*)d_in[3];
    const float* wv   = (const float*)d_in[4];
    const float* wo   = (const float*)d_in[5];

    const int Mtok = Bc * Sc;                 // 8192 token rows

    char* ws = (char*)d_ws;
    size_t off = 0;
    auto alloc = [&](size_t bytes) -> void* {
        void* p = ws + off;
        off = (off + bytes + 255) & ~(size_t)255;
        return p;
    };

    unsigned short* xn    = (unsigned short*)alloc((size_t)Mtok * Dc * 2);
    unsigned short* wqt   = (unsigned short*)alloc((size_t)Dc * Dc * 2);   // [N][K]
    unsigned short* wkt   = (unsigned short*)alloc((size_t)KVD * Dc * 2);
    unsigned short* wvt   = (unsigned short*)alloc((size_t)KVD * Dc * 2);
    unsigned short* wot   = (unsigned short*)alloc((size_t)Dc * Dc * 2);
    unsigned short* qb    = (unsigned short*)alloc((size_t)Mtok * Dc * 2);
    unsigned short* kb    = (unsigned short*)alloc((size_t)Mtok * KVD * 2);
    unsigned short* vb    = (unsigned short*)alloc((size_t)Mtok * KVD * 2);
    unsigned short* attnb = (unsigned short*)alloc((size_t)Mtok * Dc * 2);

    // 1) RMSNorm -> bf16
    rmsnorm_bf16_kernel<<<Mtok, 256, 0, stream>>>(x, ln_w, xn);

    // 2) weight transpose+convert: w[K][N] -> wt[N][K] bf16
    transpose_f32_to_bf16_kernel<<<dim3(Dc / 32, Dc / 32), 256, 0, stream>>>(
        wq, wqt, Dc, Dc);
    transpose_f32_to_bf16_kernel<<<dim3(Dc / 32, KVD / 32), 256, 0, stream>>>(
        wk, wkt, Dc, KVD);
    transpose_f32_to_bf16_kernel<<<dim3(Dc / 32, KVD / 32), 256, 0, stream>>>(
        wv, wvt, Dc, KVD);
    transpose_f32_to_bf16_kernel<<<dim3(Dc / 32, Dc / 32), 256, 0, stream>>>(
        wo, wot, Dc, Dc);

    // 3) Q/K/V projections (bf16 out)
    gemm_bf16_kernel<<<dim3(Mtok / 128, Dc / 128), 256, 0, stream>>>(
        xn, wqt, nullptr, qb, Mtok, Dc, Dc);
    gemm_bf16_kernel<<<dim3(Mtok / 128, KVD / 128), 256, 0, stream>>>(
        xn, wkt, nullptr, kb, Mtok, KVD, Dc);
    gemm_bf16_kernel<<<dim3(Mtok / 128, KVD / 128), 256, 0, stream>>>(
        xn, wvt, nullptr, vb, Mtok, KVD, Dc);

    // 4) RoPE in-place on Q and K
    {
        int qp = Mtok * Hc  * (HDc / 2);
        int kp = Mtok * KHc * (HDc / 2);
        rope_bf16_kernel<<<(qp + 255) / 256, 256, 0, stream>>>(qb, Hc,  qp);
        rope_bf16_kernel<<<(kp + 255) / 256, 256, 0, stream>>>(kb, KHc, kp);
    }

    // 5) sliding-window attention: B*H*nb*(W/128) = 2*16*8*4 = 1024 blocks
    swa_attn_kernel<<<1024, 256, 0, stream>>>(qb, kb, vb, attnb);

    // 6) output projection (fp32 out)
    gemm_bf16_kernel<<<dim3(Mtok / 128, Dc / 128), 256, 0, stream>>>(
        attnb, wot, (float*)d_out, nullptr, Mtok, Dc, Dc);
}
